// BeamConv2D_82566451298959
// MI455X (gfx1250) — matrix operand order repaired and checked by reading, not compiled, not run on hardware
//
#include <hip/hip_runtime.h>

typedef __attribute__((ext_vector_type(16))) _Float16 v16h;
typedef __attribute__((ext_vector_type(8)))  _Float16 v8h;
typedef __attribute__((ext_vector_type(8)))  float    v8f;

// Problem constants (match reference)
constexpr int N_    = 100000;
constexpr int C_    = 16;
constexpr int F_    = 16;
constexpr int W_    = 8;
constexpr int NSEG_ = 8192;

constexpr int BROWS = 16;   // rows per block for pre-conv (8 waves x 2 rows)
constexpr int BSEG  = 16;   // segments per block for post-conv

#define CAT16(lo, hi) __builtin_shufflevector((lo), (hi), 0, 1, 2, 3, 4, 5, 6, 7, \
                                              8, 9, 10, 11, 12, 13, 14, 15)
#define WMMA(a, b, c) __builtin_amdgcn_wmma_f32_16x16x32_f16(false, (a), false, (b), \
                                                             (short)0, (c), false, false)

// ---------------------------------------------------------------------------
// Prep: padded + transposed f16 weight tiles.
//   bpreT [D][F][160] : bpreT[d][f][k] = pre_w[d, k*16+f] for k<144 else 0
//   bpostT[D][F][64]  : bpostT[d][f][k] = post_w[d, k*16+f] for k<48 else 0
// ---------------------------------------------------------------------------
__global__ __launch_bounds__(256)
void pack_weights(const float* __restrict__ pre_w,    // [D,144,16]
                  const float* __restrict__ post_w,   // [D,48,16]
                  _Float16* __restrict__ bpreT,       // [D,16,160]
                  _Float16* __restrict__ bpostT)      // [D,16,64]
{
    const int t = blockIdx.x * blockDim.x + threadIdx.x;
    if (t < 3 * 16 * 160) {
        const int d = t / (16 * 160);
        const int r = t % (16 * 160);
        const int f = r / 160;
        const int k = r % 160;
        const float v = (k < 144) ? pre_w[(d * 144 + k) * 16 + f] : 0.0f;
        bpreT[t] = (_Float16)v;
    }
    const int u = t - 3 * 16 * 160;
    if (u >= 0 && u < 3 * 16 * 64) {
        const int d = u / (16 * 64);
        const int r = u % (16 * 64);
        const int f = r / 64;
        const int k = r % 64;
        const float v = (k < 48) ? post_w[(d * 48 + k) * 16 + f] : 0.0f;
        bpostT[u] = (_Float16)v;
    }
}

// ---------------------------------------------------------------------------
// Kernel 1: gather X[inv], 3x3 SAME conv (+bias, relu) via WMMA, atomic
//           scatter-add into segment buffer [NSEG, W, F].
// LDS: f16, W-halo padded: [18 rows][10 cols][16 ch]; cols 0 and 9 are zeros.
// ---------------------------------------------------------------------------
__global__ __launch_bounds__(256)
void preconv_scatter(const float* __restrict__ X,       // [N, C]
                     const int*   __restrict__ inv,     // [N, W] (this dim)
                     const int*   __restrict__ segidx,  // [N]    (this dim)
                     const _Float16* __restrict__ bT,   // [16,160] padded weights^T
                     const float* __restrict__ pb,      // [F]
                     float*       __restrict__ segbuf)  // [NSEG, W, F]
{
    __shared__ _Float16 smem[BROWS + 2][W_ + 2][C_];   // 18*10*16 f16 = 5760 B
    const int tid     = threadIdx.x;
    const int n_start = blockIdx.x * BROWS;

    // Phase 1: cooperative gather + f32->f16 convert into LDS.
    if (tid < (BROWS + 2) * W_) {                      // 144 gather threads
        const int r  = tid / W_;
        const int w  = tid % W_;
        const int gr = n_start - 1 + r;
        float vals[16];
        #pragma unroll
        for (int e = 0; e < 16; ++e) vals[e] = 0.0f;
        if (gr >= 0 && gr < N_) {
            const int idx = inv[gr * W_ + w];
            const float4* x4 = reinterpret_cast<const float4*>(X + (size_t)idx * C_);
            const float4 t0 = x4[0], t1 = x4[1], t2 = x4[2], t3 = x4[3];
            vals[0]  = t0.x; vals[1]  = t0.y; vals[2]  = t0.z; vals[3]  = t0.w;
            vals[4]  = t1.x; vals[5]  = t1.y; vals[6]  = t1.z; vals[7]  = t1.w;
            vals[8]  = t2.x; vals[9]  = t2.y; vals[10] = t2.z; vals[11] = t2.w;
            vals[12] = t3.x; vals[13] = t3.y; vals[14] = t3.z; vals[15] = t3.w;
        }
        v8h h0, h1;
        #pragma unroll
        for (int e = 0; e < 8; ++e) {
            h0[e] = (_Float16)vals[e];
            h1[e] = (_Float16)vals[8 + e];
        }
        *(v8h*)&smem[r][w + 1][0] = h0;
        *(v8h*)&smem[r][w + 1][8] = h1;
    } else if (tid < (BROWS + 2) * W_ + 2 * (BROWS + 2)) {  // 36 halo threads
        const int p   = tid - (BROWS + 2) * W_;
        const int r   = p >> 1;
        const int col = (p & 1) * (W_ + 1);
        v8h z;
        #pragma unroll
        for (int e = 0; e < 8; ++e) z[e] = (_Float16)0.0f;
        *(v8h*)&smem[r][col][0] = z;
        *(v8h*)&smem[r][col][8] = z;
    }
    __syncthreads();

    // Phase 2: each wave owns one 16-position tile (2 rows x 8 w).
    const int lane = tid & 31;
    const int wv   = tid >> 5;       // wave 0..7
    const int h    = lane >> 4;      // lane half
    const int f    = lane & 15;      // filter column (B/C/D)
    const int m    = lane & 15;      // A-matrix row M
    const int mrow = m >> 3;         // row within tile (0/1)
    const int mw   = m & 7;          // beam position

    // B fragments: e<8 <-> k = q*32 + 8h + e, e>=8 <-> k = q*32 + 16 + 8h + (e-8).
    const _Float16* bp = bT + (f * 160 + 8 * h);
    v16h b[5];
    #pragma unroll
    for (int q = 0; q < 5; ++q) {
        const v8h lo = *(const v8h*)(bp + q * 32);
        const v8h hi = *(const v8h*)(bp + q * 32 + 16);
        b[q] = CAT16(lo, hi);
    }

    const float bias = pb[f];
    v8f acc;
    #pragma unroll
    for (int j = 0; j < 8; ++j) acc[j] = bias;

    const int ldsr0 = 2 * wv + mrow;   // + dh gives LDS row (row 0 == n_start-1)
    const int c0    = 8 * h;           // channel base for this lane half

    // A tap group (compile-time dh, dw): 8 contiguous f16 -> one b128 LDS load.
    auto grp = [&](int dh, int dw) -> v8h {
        return *(const v8h*)&smem[ldsr0 + dh][mw + dw][c0];
    };
    v8h z8;
    #pragma unroll
    for (int e = 0; e < 8; ++e) z8[e] = (_Float16)0.0f;

    // chunk q covers taps 2q, 2q+1 (tap = 3*dh + dw); tap 9 is K-padding.
    acc = WMMA(CAT16(grp(0, 0), grp(0, 1)), b[0], acc);
    acc = WMMA(CAT16(grp(0, 2), grp(1, 0)), b[1], acc);
    acc = WMMA(CAT16(grp(1, 1), grp(1, 2)), b[2], acc);
    acc = WMMA(CAT16(grp(2, 0), grp(2, 1)), b[3], acc);
    acc = WMMA(CAT16(grp(2, 2), z8),        b[4], acc);

    // relu + segment scatter-add. C/D layout: VGPR j <-> M = j + 8h,
    // so lanes 0-15 write tile-row 0, lanes 16-31 write tile-row 1; w = j.
    const int row = n_start + 2 * wv + h;
    const int seg = segidx[row];
    #pragma unroll
    for (int j = 0; j < 8; ++j) {
        const float v = acc[j] > 0.0f ? acc[j] : 0.0f;
        atomicAdd(&segbuf[(seg * W_ + j) * F_ + f], v);
    }
}

// ---------------------------------------------------------------------------
// Kernel 2: (3,1) SAME conv over the segment axis (+bias, relu) via WMMA.
// LDS: f16, [18 seg rows][8 w][16 ch] flat.
// ---------------------------------------------------------------------------
__global__ __launch_bounds__(256)
void postconv(const float* __restrict__ segbuf,     // [NSEG, W, F]
              const _Float16* __restrict__ bT,      // [16,64] padded weights^T
              const float* __restrict__ pb,         // [F]
              float*       __restrict__ postbuf)    // [NSEG, W, F]
{
    __shared__ _Float16 smem[(BSEG + 2) * W_ * F_];  // 2304 f16 = 4608 B
    const int tid     = threadIdx.x;
    const int s_start = blockIdx.x * BSEG;

    // Fill: float4 granules, convert to f16, 8-byte LDS stores.
    typedef __attribute__((ext_vector_type(4))) _Float16 v4h;
    for (int i = tid; i < (BSEG + 2) * W_ * F_ / 4; i += 256) {   // 576 float4
        const int r  = i >> 5;                 // 32 float4 per seg row
        const int gs = s_start - 1 + r;
        float4 v = {0.0f, 0.0f, 0.0f, 0.0f};
        if (gs >= 0 && gs < NSEG_)
            v = reinterpret_cast<const float4*>(segbuf)[gs * 32 + (i & 31)];
        v4h hv;
        hv[0] = (_Float16)v.x; hv[1] = (_Float16)v.y;
        hv[2] = (_Float16)v.z; hv[3] = (_Float16)v.w;
        *(v4h*)&smem[i * 4] = hv;
    }
    __syncthreads();

    const int lane = tid & 31;
    const int wv   = tid >> 5;
    const int h    = lane >> 4;
    const int f    = lane & 15;
    const int m    = lane & 15;
    const int mrow = m >> 3;
    const int mw   = m & 7;

    const _Float16* bp = bT + (f * 64 + 8 * h);
    v16h b[2];
    #pragma unroll
    for (int q = 0; q < 2; ++q) {
        const v8h lo = *(const v8h*)(bp + q * 32);
        const v8h hi = *(const v8h*)(bp + q * 32 + 16);
        b[q] = CAT16(lo, hi);
    }

    const float bias = pb[f];
    v8f acc;
    #pragma unroll
    for (int j = 0; j < 8; ++j) acc[j] = bias;

    const int ldsr0 = 2 * wv + mrow;
    const int c0    = 8 * h;
    auto grp = [&](int ds) -> v8h {
        return *(const v8h*)&smem[(ldsr0 + ds) * (W_ * F_) + mw * F_ + c0];
    };
    v8h z8;
    #pragma unroll
    for (int e = 0; e < 8; ++e) z8[e] = (_Float16)0.0f;

    acc = WMMA(CAT16(grp(0), grp(1)), b[0], acc);   // taps ds=0,1
    acc = WMMA(CAT16(grp(2), z8),     b[1], acc);   // tap ds=2 + K-padding

    const int s = s_start + 2 * wv + h;
    #pragma unroll
    for (int j = 0; j < 8; ++j) {
        const float v = acc[j] > 0.0f ? acc[j] : 0.0f;
        postbuf[(s * W_ + j) * F_ + f] = v;
    }
}

// ---------------------------------------------------------------------------
// Kernel 3: Y[n, :, :] += postbuf[segidx[n], :, :]  (vectorized float4)
// ---------------------------------------------------------------------------
__global__ __launch_bounds__(256)
void gather_add(const float4* __restrict__ postbuf,  // [NSEG * 32] float4
                const int*    __restrict__ segidx,   // [N]
                float4*       __restrict__ Y)        // [N * 32] float4
{
    const int t = blockIdx.x * blockDim.x + threadIdx.x;
    if (t >= N_ * 32) return;
    const int n   = t >> 5;
    const int r   = t & 31;
    const int seg = segidx[n];
    float4 a = Y[t];
    const float4 p = postbuf[seg * 32 + r];
    a.x += p.x; a.y += p.y; a.z += p.z; a.w += p.w;
    Y[t] = a;
}

// ---------------------------------------------------------------------------
extern "C" void kernel_launch(void* const* d_in, const int* in_sizes, int n_in,
                              void* d_out, int out_size, void* d_ws, size_t ws_size,
                              hipStream_t stream) {
    const float* X      = (const float*)d_in[0];
    const int*   inv    = (const int*)  d_in[1];
    const int*   index  = (const int*)  d_in[2];
    const float* pre_w  = (const float*)d_in[3];
    const float* pre_b  = (const float*)d_in[4];
    const float* post_w = (const float*)d_in[5];
    const float* post_b = (const float*)d_in[6];
    float* Y = (float*)d_out;

    float* segbuf  = (float*)d_ws;                        // 4 MB
    float* postbuf = segbuf + (size_t)NSEG_ * W_ * F_;    // 4 MB
    _Float16* bpreT  = (_Float16*)(postbuf + (size_t)NSEG_ * W_ * F_);
    _Float16* bpostT = bpreT + 3 * 16 * 160;

    hipMemsetAsync(d_out, 0, (size_t)N_ * W_ * F_ * sizeof(float), stream);

    // Pack padded/transposed f16 weights (3*2560 + 3*1024 = 10752 elements).
    pack_weights<<<42, 256, 0, stream>>>(pre_w, post_w, bpreT, bpostT);

    for (int d = 0; d < 3; ++d) {
        hipMemsetAsync(segbuf, 0, (size_t)NSEG_ * W_ * F_ * sizeof(float), stream);
        preconv_scatter<<<N_ / BROWS, 256, 0, stream>>>(
            X,
            inv    + (size_t)d * N_ * W_,
            index  + (size_t)d * N_,
            bpreT  + (size_t)d * 16 * 160,
            pre_b  + (size_t)d * F_,
            segbuf);
        postconv<<<NSEG_ / BSEG, 256, 0, stream>>>(
            segbuf,
            bpostT + (size_t)d * 16 * 64,
            post_b + (size_t)d * F_,
            postbuf);
        gather_add<<<(N_ * 32 + 255) / 256, 256, 0, stream>>>(
            (const float4*)postbuf,
            index + (size_t)d * N_,
            (float4*)Y);
    }
}
